// LigerLMHeadDPO_58059367907835
// MI455X (gfx1250) — matrix operand order repaired and checked by reading, not compiled, not run on hardware
//
#include <hip/hip_runtime.h>
#include <hip/hip_bf16.h>

// ---------------- problem constants (from reference) ----------------
#define B2 8
#define TT 512
#define HH 2048
#define VV 32000
#define NTOK (B2 * TT)          // 4096 tokens
#define IGNORE_IDX (-100)
#define BETA 0.1f

typedef __attribute__((ext_vector_type(16))) __bf16 bf16x16;
typedef __attribute__((ext_vector_type(8)))  __bf16 bf16x8;
typedef __attribute__((ext_vector_type(8)))  float  f32x8;

// raw hardware v_exp_f32 (base-2); args here are bounded, no clamp scaffolding
__device__ __forceinline__ float exp2_raw(float x) { return __builtin_amdgcn_exp2f(x); }

// ---------------- fp32 -> bf16 conversion (vector of 8) ----------------
__global__ __launch_bounds__(256) void cvt_bf16_kernel(const float* __restrict__ src,
                                                       __bf16* __restrict__ dst, int n8) {
    int i = blockIdx.x * 256 + threadIdx.x;
    if (i >= n8) return;
    const float4* s = (const float4*)src;
    float4 a = s[(size_t)i * 2 + 0];
    float4 b = s[(size_t)i * 2 + 1];
    bf16x8 o;
    o[0] = (__bf16)a.x; o[1] = (__bf16)a.y; o[2] = (__bf16)a.z; o[3] = (__bf16)a.w;
    o[4] = (__bf16)b.x; o[5] = (__bf16)b.y; o[6] = (__bf16)b.z; o[7] = (__bf16)b.w;
    *(bf16x8*)(dst + (size_t)i * 8) = o;
}

// ---------------- flash-logsumexp GEMM via bf16 WMMA ----------------
// One block = 16 tokens (M tile). 8 waves; each wave owns TWO 16-wide N slices
// per vocab iteration (block covers N=256/iter, 125 iters over V=32000).
// Logits never touch memory: per-lane online logsumexp in base-2 over the
// WMMA f32 accumulators, final LDS combine.
__global__ __launch_bounds__(256) void lse_kernel(const __bf16* __restrict__ xb,  // [NTOK, HH]
                                                  const __bf16* __restrict__ wb,  // [VV, HH]
                                                  float* __restrict__ lse)        // [NTOK]
{
    __shared__ union {
        __bf16 xt[16][HH];          // 64 KB x tile (whole K for 16 tokens)
        float  red[8][32][8][2];    // 16 KB reduction scratch (reused after sweep)
    } sm;

    const int tid  = threadIdx.x;
    const int wave = tid >> 5;
    const int lane = tid & 31;
    const int l15  = lane & 15;
    const int hi   = lane >> 4;
    const int mbase = blockIdx.x * 16;

    // stage x tile: 16 rows x 2048 bf16 = 4096 uint4
    {
        const uint4* s = (const uint4*)(xb + (size_t)mbase * HH);
        uint4* d = (uint4*)&sm.xt[0][0];
        for (int i = tid; i < 4096; i += 256) d[i] = s[i];
    }
    __syncthreads();

    // A operand (16-bit 16x32): lane<16 -> row l15, K chunks {0..7,16..23};
    // lane>=16 -> row l15, K chunks {8..15,24..31}
    const __bf16* arow = &sm.xt[l15][hi ? 8 : 0];
    const int woff = hi ? 8 : 0;        // B: lanes 0-15 hold K {0..7,16..23}, 16-31 hold {8..15,24..31}

    const float LOG2E = 1.44269504088896340736f;
    float rmax[8], rsum[8];
#pragma unroll
    for (int g = 0; g < 8; ++g) { rmax[g] = -3.0e38f; rsum[g] = 0.0f; }

    for (int nb = wave * 16; nb < VV; nb += 256) {
        // two B columns per wave: nb and nb+128
        const __bf16* wrow0 = wb + (size_t)(nb + l15) * HH + woff;
        const __bf16* wrow1 = wb + (size_t)(nb + 128 + l15) * HH + woff;

        f32x8 acc0 = {};
        f32x8 acc1 = {};
#pragma unroll 2
        for (int k0 = 0; k0 < HH; k0 += 32) {
            union { bf16x16 v; bf16x8 h[2]; } a, b0, b1;
            a.h[0]  = *(const bf16x8*)(arow + k0);
            a.h[1]  = *(const bf16x8*)(arow + k0 + 16);
            b0.h[0] = *(const bf16x8*)(wrow0 + k0);
            b0.h[1] = *(const bf16x8*)(wrow0 + k0 + 16);
            b1.h[0] = *(const bf16x8*)(wrow1 + k0);
            b1.h[1] = *(const bf16x8*)(wrow1 + k0 + 16);
            acc0 = __builtin_amdgcn_wmma_f32_16x16x32_bf16(
                false, a.v, false, b0.v, (short)0, acc0, false, false);
            acc1 = __builtin_amdgcn_wmma_f32_16x16x32_bf16(
                false, a.v, false, b1.v, (short)0, acc1, false, false);
        }
        // online logsumexp (base-2, raw v_exp_f32). VGPR g holds row g
        // (lanes 0-15) / row g+8 (lanes 16-31); both tiles fold into the same
        // per-lane running state (reduction over all n this lane sees).
#pragma unroll
        for (int g = 0; g < 8; ++g) {
            float v0 = acc0[g] * LOG2E;
            float v1 = acc1[g] * LOG2E;
            float nm = fmaxf(rmax[g], fmaxf(v0, v1));
            rsum[g]  = rsum[g] * exp2_raw(rmax[g] - nm)
                     + exp2_raw(v0 - nm) + exp2_raw(v1 - nm);
            rmax[g]  = nm;
        }
    }

    __syncthreads();   // all LDS x-tile reads done before union reuse
#pragma unroll
    for (int g = 0; g < 8; ++g) {
        sm.red[wave][lane][g][0] = rmax[g];
        sm.red[wave][lane][g][1] = rsum[g];
    }
    __syncthreads();

    if (tid < 16) {                 // one thread per token row
        const int m     = tid;
        const int g     = m & 7;
        const int lbase = (m >> 3) ? 16 : 0;
        float fm = -3.0e38f, fs = 0.0f;
        for (int w = 0; w < 8; ++w)
            for (int l = 0; l < 16; ++l) {
                float pm = sm.red[w][lbase + l][g][0];
                float ps = sm.red[w][lbase + l][g][1];
                float nm = fmaxf(fm, pm);
                fs = fs * exp2_raw(fm - nm) + ps * exp2_raw(pm - nm);
                fm = nm;
            }
        lse[mbase + m] = 0.69314718055994530942f * (fm + log2f(fs));
    }
}

// ---------------- exact fp32 target-token logit ----------------
// one wave per token: dot(x_t, W[y_t]) over H=2048
__global__ __launch_bounds__(256) void tgt_kernel(const float* __restrict__ x,
                                                  const float* __restrict__ w,
                                                  const long long* __restrict__ y,
                                                  float* __restrict__ tgt) {
    const int token = blockIdx.x * 8 + (threadIdx.x >> 5);
    const int lane  = threadIdx.x & 31;
    if (token >= NTOK) return;
    long long yv = y[token];
    float s = 0.0f;
    if (yv >= 0 && yv < VV) {
        const float* xr = x + (size_t)token * HH;
        const float* wr = w + (size_t)yv * HH;
        for (int k = lane * 4; k < HH; k += 128) {
            float4 xv = *(const float4*)(xr + k);
            float4 wv = *(const float4*)(wr + k);
            s += xv.x * wv.x + xv.y * wv.y + xv.z * wv.z + xv.w * wv.w;
        }
    }
    for (int off = 16; off > 0; off >>= 1) s += __shfl_down(s, off, 32);
    if (lane == 0) tgt[token] = s;
}

// ---------------- final DPO reduction ----------------
__global__ __launch_bounds__(256) void loss_kernel(const long long* __restrict__ y,
                                                   const float* __restrict__ tgt_pol,
                                                   const float* __restrict__ lse_pol,
                                                   const float* __restrict__ tgt_ref,
                                                   const float* __restrict__ lse_ref,
                                                   float* __restrict__ out) {
    __shared__ float pol_s[B2], ref_s[B2];
    __shared__ int   cnt[B2];
    const int tid = threadIdx.x;
    if (tid < B2) { pol_s[tid] = 0.0f; ref_s[tid] = 0.0f; cnt[tid] = 0; }
    __syncthreads();
    for (int t = tid; t < NTOK; t += 256) {
        int b = t >> 9;                       // /512
        if (y[t] != IGNORE_IDX) {
            atomicAdd(&pol_s[b], tgt_pol[t] - lse_pol[t]);
            atomicAdd(&ref_s[b], tgt_ref[t] - lse_ref[t]);
            atomicAdd(&cnt[b], 1);
        }
    }
    __syncthreads();
    if (tid == 0) {
        const int B = B2 / 2;
        float loss = 0.0f;
        for (int p = 0; p < B; ++p) {
            float nc = (float)max(cnt[p], 1),     nr = (float)max(cnt[p + B], 1);
            float pc = pol_s[p] / nc,             pr = pol_s[p + B] / nr;
            float rc = ref_s[p] / nc,             rr = ref_s[p + B] / nr;
            float z  = BETA * ((pc - rc) - (pr - rr));
            // log_sigmoid(z) = min(z,0) - log1p(exp(-|z|))
            loss -= fminf(z, 0.0f) - log1pf(expf(-fabsf(z)));
        }
        out[0] = loss / (float)B;
    }
}

// ---------------- launch ----------------
extern "C" void kernel_launch(void* const* d_in, const int* in_sizes, int n_in,
                              void* d_out, int out_size, void* d_ws, size_t ws_size,
                              hipStream_t stream) {
    const float*     x     = (const float*)d_in[0];
    const float*     ref_x = (const float*)d_in[1];
    const long long* y     = (const long long*)d_in[2];
    const float*     W     = (const float*)d_in[3];
    const float*     ref_W = (const float*)d_in[4];
    float* out = (float*)d_out;

    // workspace layout (bf16 W buffer reused sequentially for policy/ref)
    char* ws = (char*)d_ws;
    const size_t W_BYTES = (size_t)VV * HH * 2;          // 131,072,000
    const size_t X_BYTES = (size_t)NTOK * HH * 2;        //  16,777,216
    __bf16* wb = (__bf16*)ws;
    __bf16* xb = (__bf16*)(ws + W_BYTES);
    float* lse_pol = (float*)(ws + W_BYTES + X_BYTES);
    float* lse_ref = lse_pol + NTOK;
    float* tgt_pol = lse_ref + NTOK;
    float* tgt_ref = tgt_pol + NTOK;

    const int nW8 = (VV * HH) / 8;       // 8,192,000
    const int nX8 = (NTOK * HH) / 8;     // 1,048,576

    // ---- policy pass ----
    cvt_bf16_kernel<<<(nW8 + 255) / 256, 256, 0, stream>>>(W, wb, nW8);
    cvt_bf16_kernel<<<(nX8 + 255) / 256, 256, 0, stream>>>(x, xb, nX8);
    lse_kernel<<<NTOK / 16, 256, 0, stream>>>(xb, wb, lse_pol);
    tgt_kernel<<<NTOK / 8, 256, 0, stream>>>(x, W, y, tgt_pol);

    // ---- reference pass (reuses wb/xb; stream order guarantees safety) ----
    cvt_bf16_kernel<<<(nW8 + 255) / 256, 256, 0, stream>>>(ref_W, wb, nW8);
    cvt_bf16_kernel<<<(nX8 + 255) / 256, 256, 0, stream>>>(ref_x, xb, nX8);
    lse_kernel<<<NTOK / 16, 256, 0, stream>>>(xb, wb, lse_ref);
    tgt_kernel<<<NTOK / 8, 256, 0, stream>>>(ref_x, ref_W, y, tgt_ref);

    // ---- DPO loss ----
    loss_kernel<<<1, 256, 0, stream>>>(y, tgt_pol, lse_pol, tgt_ref, lse_ref, out);
}
